// GMMBlock_36197984371027
// MI455X (gfx1250) — matrix operand order repaired and checked by reading, not compiled, not run on hardware
//
#include <hip/hip_runtime.h>

// ---------------------------------------------------------------------------
// Types / WMMA helpers (CDNA5 gfx1250, wave32, v_wmma_f32_16x16x32_f16)
// ---------------------------------------------------------------------------
typedef __attribute__((ext_vector_type(8)))  _Float16 v8h;
typedef __attribute__((ext_vector_type(16))) _Float16 v16h;
typedef __attribute__((ext_vector_type(8)))  float    v8f;

static __device__ __forceinline__ v16h cat16(v8h a, v8h b) {
    return __builtin_shufflevector(a, b, 0,1,2,3,4,5,6,7,8,9,10,11,12,13,14,15);
}

// A fragment (16x32 f16, M x K). ISA layout: lane = (m = lane&15, hi = lane>>4),
// VGPR0..3 hold K = hi*8 .. hi*8+7, VGPR4..7 hold K = 16+hi*8 .. 16+hi*8+7.
static __device__ __forceinline__ v16h load_a_frag(const _Float16* base, int ld) {
    int lane = threadIdx.x & 31;
    int m = lane & 15, hi = lane >> 4;
    v8h lo = *(const v8h*)(base + m * ld + hi * 8);
    v8h hh = *(const v8h*)(base + m * ld + 16 + hi * 8);
    return cat16(lo, hh);
}

// B fragment (32x16 f16, K x N) from a column-major staging Bt[n][k] (k contiguous).
// ISA layout: lane = (n = lane&15, hi = lane>>4), elements 0..7 hold K = hi*8 .. hi*8+7, 8..15 hold K = 16+hi*8 .. 16+hi*8+7.
static __device__ __forceinline__ v16h load_b_frag(const _Float16* baseT, int ld) {
    int lane = threadIdx.x & 31;
    int n = lane & 15, hi = lane >> 4;
    v8h b0 = *(const v8h*)(baseT + n * ld + hi * 8);
    v8h b1 = *(const v8h*)(baseT + n * ld + 16 + hi * 8);
    return cat16(b0, b1);
}

static __device__ __forceinline__ v8f wmma_f16(v16h a, v16h b, v8f c) {
    return __builtin_amdgcn_wmma_f32_16x16x32_f16(false, a, false, b, (short)0, c,
                                                  false, false);
}

// ---------------------------------------------------------------------------
// CDNA5 async global->LDS DMA (no VGPR round-trip, tracked with ASYNCcnt)
// ---------------------------------------------------------------------------
static __device__ __forceinline__ void async_ld_b128(_Float16* lds_dst,
                                                     const _Float16* gsrc) {
    unsigned loff = (unsigned)(unsigned long long)lds_dst;  // low 32b = LDS offset
    asm volatile("global_load_async_to_lds_b128 %0, %1, off"
                 :: "v"(loff), "v"(gsrc)
                 : "memory");
}
static __device__ __forceinline__ void wait_async0() {
    asm volatile("s_wait_asynccnt 0" ::: "memory");
}

// ---------------------------------------------------------------------------
// Problem constants
// ---------------------------------------------------------------------------
#define NB   4
#define LSEQ 1024
#define HH   12
#define DHD  64
#define DD   768
#define MROWS (NB * LSEQ)          // 4096
#define DSQ   (DD * DD)            // 589824

// Workspace layout (bytes, all 256-aligned)
#define XH_OFF     0u
#define WQKVT_OFF  6291456u        // 3*D*D f16
#define WOUTT_OFF  9830400u        // D*D f16
#define QH_OFF     11010048u       // M*D f16 (head layout)
#define KH_OFF     17301504u
#define VT_OFF     23592960u       // transposed [nb,h,dh,l]
#define CTX_OFF    29884416u       // M*D f16
#define RESID_OFF  36175872u       // M*D f32

// ---------------------------------------------------------------------------
// Elementwise converts
// ---------------------------------------------------------------------------
__global__ void __launch_bounds__(256) cvt_x_kernel(const float* __restrict__ x,
                                                    _Float16* __restrict__ xh, int n) {
    int i = blockIdx.x * 256 + threadIdx.x;
    if (i < n) xh[i] = (_Float16)x[i];
}

// Convert one branch's weights to f16, transposed: Wt[mat][n][k] = W[mat][k][n]
__global__ void __launch_bounds__(256) cvt_w_kernel(const float* __restrict__ qkvw,
                                                    const float* __restrict__ outw,
                                                    _Float16* __restrict__ wqkvt,
                                                    _Float16* __restrict__ woutt) {
    int i = blockIdx.x * 256 + threadIdx.x;     // over 4 * D*D
    int mat = i / DSQ;
    int r   = i % DSQ;
    int k = r / DD, n = r % DD;
    float v = (mat < 3) ? qkvw[mat * DSQ + k * DD + n] : outw[k * DD + n];
    if (mat < 3) wqkvt[mat * DSQ + n * DD + k] = (_Float16)v;
    else         woutt[n * DD + k]             = (_Float16)v;
}

// ---------------------------------------------------------------------------
// Shared WMMA mainloop: C[128x64] += A[128x768] * B[768x64]
// Double-buffered LDS, async global->LDS staging, 8 waves = 4(M) x 2(N),
// each wave owns a 32x32 tile = 2x2 wmma fragments (4 wmma / K-step).
// ---------------------------------------------------------------------------
#define ALD 40           // 32 + 8 f16 pad: 80B rows (16B aligned, bank-rotating)
#define BLD 40
#define ASZ (128 * ALD)  // f16 elements per A buffer
#define BSZ (64 * BLD)

static __device__ __forceinline__ void gemm_128x64(
        const _Float16* __restrict__ Ag,   // A rows, row-major, ld = DD
        const _Float16* __restrict__ Bt,   // B cols (n-major, k contiguous), ld = DD
        _Float16* As, _Float16* Bs,        // LDS double buffers
        v8f& c00, v8f& c01, v8f& c10, v8f& c11) {
    int tid = threadIdx.x;
    int w = tid >> 5;
    int mq = w >> 1, nq = w & 1;
    // A staging: 512 b128 chunks (row = chunk>>2, q = chunk&3), 2 per thread
    int ar = tid >> 1, aq0 = (tid & 1) * 2;
    // B staging: 256 b128 chunks, 1 per thread
    int br = tid >> 2, bq = tid & 3;

    // preload K-slab 0
    async_ld_b128(As + ar * ALD + aq0 * 8,       Ag + ar * DD + aq0 * 8);
    async_ld_b128(As + ar * ALD + (aq0 + 1) * 8, Ag + ar * DD + (aq0 + 1) * 8);
    async_ld_b128(Bs + br * BLD + bq * 8,        Bt + br * DD + bq * 8);
    wait_async0();
    __syncthreads();

    for (int it = 0; it < DD / 32; ++it) {
        int cur = it & 1;
        if (it + 1 < DD / 32) {
            int nb = cur ^ 1, kk = (it + 1) * 32;
            async_ld_b128(As + nb * ASZ + ar * ALD + aq0 * 8,
                          Ag + ar * DD + kk + aq0 * 8);
            async_ld_b128(As + nb * ASZ + ar * ALD + (aq0 + 1) * 8,
                          Ag + ar * DD + kk + (aq0 + 1) * 8);
            async_ld_b128(Bs + nb * BSZ + br * BLD + bq * 8,
                          Bt + br * DD + kk + bq * 8);
            if (it + 2 < DD / 32) {
                __builtin_prefetch(Ag + ar * DD + (it + 2) * 32, 0, 1);
                __builtin_prefetch(Bt + br * DD + (it + 2) * 32, 0, 1);
            }
        }
        const _Float16* Ab = As + cur * ASZ + (mq * 32) * ALD;
        const _Float16* Bb = Bs + cur * BSZ + (nq * 32) * BLD;
        v16h a0 = load_a_frag(Ab, ALD);
        v16h a1 = load_a_frag(Ab + 16 * ALD, ALD);
        v16h b0 = load_b_frag(Bb, BLD);
        v16h b1 = load_b_frag(Bb + 16 * BLD, BLD);
        c00 = wmma_f16(a0, b0, c00);
        c01 = wmma_f16(a0, b1, c01);
        c10 = wmma_f16(a1, b0, c10);
        c11 = wmma_f16(a1, b1, c11);
        wait_async0();
        __syncthreads();
    }
}

// ---------------------------------------------------------------------------
// QKV GEMM: [4096x768] x [768x768] (x3 matrices).
// Epilogue writes Q(*0.125)/K in [nb,h,l,dh], V transposed [nb,h,dh,l].
// ---------------------------------------------------------------------------
__global__ void __launch_bounds__(256) qkv_gemm_kernel(
        const _Float16* __restrict__ Xh, const _Float16* __restrict__ Wt,
        const float* __restrict__ bias3, _Float16* __restrict__ Qh,
        _Float16* __restrict__ Kh, _Float16* __restrict__ Vt) {
    __shared__ alignas(16) _Float16 As[2 * ASZ];
    __shared__ alignas(16) _Float16 Bs[2 * BSZ];

    int m0  = blockIdx.x * 128;
    int gy  = blockIdx.y;
    int j   = gy / 12;               // 0=Q 1=K 2=V
    int nn0 = (gy % 12) * 64;

    v8f c00 = {}, c01 = {}, c10 = {}, c11 = {};
    gemm_128x64(Xh + m0 * DD, Wt + j * DSQ + nn0 * DD, As, Bs, c00, c01, c10, c11);

    int tid = threadIdx.x, w = tid >> 5, lane = tid & 31;
    int mq = w >> 1, nq = w & 1;
    int colL = lane & 15, hic = lane >> 4;
    v8f cc[2][2] = {{c00, c01}, {c10, c11}};
#pragma unroll
    for (int mm = 0; mm < 2; ++mm) {
#pragma unroll
        for (int nn = 0; nn < 2; ++nn) {
            v8f c = cc[mm][nn];
            int ngl = nn0 + nq * 32 + nn * 16 + colL;
            float bv = bias3[j * DD + ngl];
            int h = ngl >> 6, dh = ngl & 63;
#pragma unroll
            for (int r = 0; r < 8; ++r) {
                int mg = m0 + mq * 32 + mm * 16 + r + 8 * hic;
                int nb = mg >> 10, lpos = mg & 1023;
                float v = c[r] + bv;
                if (j == 0)
                    Qh[((nb * HH + h) * LSEQ + lpos) * DHD + dh] =
                        (_Float16)(v * 0.125f);
                else if (j == 1)
                    Kh[((nb * HH + h) * LSEQ + lpos) * DHD + dh] = (_Float16)v;
                else
                    Vt[((nb * HH + h) * DHD + dh) * LSEQ + lpos] = (_Float16)v;
            }
        }
    }
}

// ---------------------------------------------------------------------------
// Attention: one block per (nb,h, 16-query tile). Full score row (16x1024)
// in LDS (f16, padded stride). Gaussian mask computed analytically:
//   g_norm(q,k) = exp(-((pos_k-center_q)^2 - dmin_q^2) * inv2w2)
// ---------------------------------------------------------------------------
__global__ void __launch_bounds__(256) attn_kernel(
        const _Float16* __restrict__ Qh, const _Float16* __restrict__ Kh,
        const _Float16* __restrict__ Vt, const float* __restrict__ mask,
        _Float16* __restrict__ ctxh, float inv2w2, int use_gauss) {
    constexpr int SLD = 1032;  // 1024 + 8 f16 pad (2064B, 16B aligned)
    __shared__ alignas(16) _Float16 S[16 * SLD];
    __shared__ float red[256];
    __shared__ float ctxacc[16 * 64];

    int q0 = blockIdx.x * 16;
    int nb = blockIdx.y / HH, h = blockIdx.y % HH;
    const _Float16* Qp = Qh + ((nb * HH + h) * LSEQ + q0) * DHD;
    const _Float16* Kp = Kh + ((nb * HH + h) * LSEQ) * DHD;
    const _Float16* Vp = Vt + ((nb * HH + h) * DHD) * LSEQ;
    const float*    mp = mask + nb * LSEQ * LSEQ + q0 * LSEQ;

    int tid = threadIdx.x, w = tid >> 5, lane = tid & 31;
    int colL = lane & 15, hic = lane >> 4;

    // Q fragments (reused for all key tiles)
    v16h aq0 = load_a_frag(Qp, DHD);       // dh 0..31
    v16h aq1 = load_a_frag(Qp + 32, DHD);  // dh 32..63

    // ---- scores: each wave owns key tiles w, w+8, ... (64 tiles total) ----
    for (int kt = w; kt < 64; kt += 8) {
        int key0 = kt * 16;
        v16h bk0 = load_b_frag(Kp + key0 * DHD, DHD);
        v16h bk1 = load_b_frag(Kp + key0 * DHD + 32, DHD);
        v8f c = {};
        c = wmma_f16(aq0, bk0, c);
        c = wmma_f16(aq1, bk1, c);

        int key = key0 + colL;
        float pos = (float)key * (1.0f / 1023.0f);
#pragma unroll
        for (int r = 0; r < 8; ++r) {
            int qloc = r + 8 * hic;
            float s = c[r];
            if (use_gauss) {
                float center = (float)(q0 + qloc) * (1.0f / 1024.0f);
                float d = pos - center;
                float kstar = rintf(center * 1023.0f);
                float dmin = kstar * (1.0f / 1023.0f) - center;
                s *= __expf(-(d * d - dmin * dmin) * inv2w2);
            }
            s += (1.0f - mp[qloc * LSEQ + key]) * -10000.0f;
            S[qloc * SLD + key] = (_Float16)s;
        }
    }
    __syncthreads();

    // ---- softmax over each of the 16 rows (16 segments per row) ----
    int row = tid >> 4, seg = tid & 15;
    float mx = -1e30f;
    for (int e = seg; e < LSEQ; e += 16) mx = fmaxf(mx, (float)S[row * SLD + e]);
    red[tid] = mx;
    __syncthreads();
    float rm = -1e30f;
    for (int i = 0; i < 16; ++i) rm = fmaxf(rm, red[row * 16 + i]);
    float sm = 0.0f;
    for (int e = seg; e < LSEQ; e += 16) sm += __expf((float)S[row * SLD + e] - rm);
    __syncthreads();
    red[tid] = sm;
    __syncthreads();
    float sum = 0.0f;
    for (int i = 0; i < 16; ++i) sum += red[row * 16 + i];
    float inv = 1.0f / sum;
    for (int e = seg; e < LSEQ; e += 16)
        S[row * SLD + e] = (_Float16)(__expf((float)S[row * SLD + e] - rm) * inv);
    for (int i = tid; i < 16 * 64; i += 256) ctxacc[i] = 0.0f;
    __syncthreads();

    // ---- ctx = P @ V : 8 waves = 4 dh-tiles x 2 key halves ----
    int dhtile = w & 3, khalf = w >> 2;
    v8f cc = {};
    for (int ks = khalf * 512; ks < khalf * 512 + 512; ks += 32) {
        v16h a = load_a_frag(S + ks, SLD);
        v16h b = load_b_frag(Vp + dhtile * 16 * LSEQ + ks, LSEQ);
        cc = wmma_f16(a, b, cc);
    }
#pragma unroll
    for (int r = 0; r < 8; ++r)
        atomicAdd(&ctxacc[(r + 8 * hic) * 64 + dhtile * 16 + colL], cc[r]);
    __syncthreads();

    for (int i = tid; i < 16 * 64; i += 256) {
        int q = i >> 6, dh = i & 63;
        ctxh[((nb << 10) + q0 + q) * DD + h * DHD + dh] = (_Float16)ctxacc[i];
    }
}

// ---------------------------------------------------------------------------
// Output projection: resid = ctx @ Wout + b + x   (f32 result for LN)
// ---------------------------------------------------------------------------
__global__ void __launch_bounds__(256) out_gemm_kernel(
        const _Float16* __restrict__ Ah, const _Float16* __restrict__ Wt,
        const float* __restrict__ bias, const float* __restrict__ X,
        float* __restrict__ R) {
    __shared__ alignas(16) _Float16 As[2 * ASZ];
    __shared__ alignas(16) _Float16 Bs[2 * BSZ];

    int m0  = blockIdx.x * 128;
    int nn0 = blockIdx.y * 64;

    v8f c00 = {}, c01 = {}, c10 = {}, c11 = {};
    gemm_128x64(Ah + m0 * DD, Wt + nn0 * DD, As, Bs, c00, c01, c10, c11);

    int tid = threadIdx.x, w = tid >> 5, lane = tid & 31;
    int mq = w >> 1, nq = w & 1;
    int colL = lane & 15, hic = lane >> 4;
    v8f cc[2][2] = {{c00, c01}, {c10, c11}};
#pragma unroll
    for (int mm = 0; mm < 2; ++mm) {
#pragma unroll
        for (int nn = 0; nn < 2; ++nn) {
            v8f c = cc[mm][nn];
            int ng = nn0 + nq * 32 + nn * 16 + colL;
            float bv = bias[ng];
#pragma unroll
            for (int r = 0; r < 8; ++r) {
                int mg = m0 + mq * 32 + mm * 16 + r + 8 * hic;
                R[mg * DD + ng] = c[r] + bv + X[mg * DD + ng];
            }
        }
    }
}

// ---------------------------------------------------------------------------
// LayerNorm + branch-average accumulate into d_out (float)
// ---------------------------------------------------------------------------
__global__ void __launch_bounds__(256) ln_kernel(
        const float* __restrict__ R, const float* __restrict__ g,
        const float* __restrict__ b, float* __restrict__ out, int first) {
    __shared__ float rs[256], rss[256];
    __shared__ float mu_s, rsig_s;
    int m = blockIdx.x, t = threadIdx.x;
    float s = 0.0f, ss = 0.0f;
    for (int e = t; e < DD; e += 256) {
        float v = R[m * DD + e];
        s += v; ss += v * v;
    }
    rs[t] = s; rss[t] = ss;
    __syncthreads();
    for (int o = 128; o > 0; o >>= 1) {
        if (t < o) { rs[t] += rs[t + o]; rss[t] += rss[t + o]; }
        __syncthreads();
    }
    if (t == 0) {
        float mu = rs[0] * (1.0f / DD);
        float var = rss[0] * (1.0f / DD) - mu * mu;
        mu_s = mu;
        rsig_s = rsqrtf(var + 1e-5f);
    }
    __syncthreads();
    float mu = mu_s, rsig = rsig_s;
    for (int e = t; e < DD; e += 256) {
        float y = (R[m * DD + e] - mu) * rsig * g[e] + b[e];
        float o25 = 0.25f * y;
        if (first) out[m * DD + e] = o25;
        else       out[m * DD + e] += o25;
    }
}

// ---------------------------------------------------------------------------
// Host launcher
// ---------------------------------------------------------------------------
extern "C" void kernel_launch(void* const* d_in, const int* in_sizes, int n_in,
                              void* d_out, int out_size, void* d_ws, size_t ws_size,
                              hipStream_t stream) {
    (void)in_sizes; (void)n_in; (void)out_size; (void)ws_size;
    const float* x    = (const float*)d_in[0];
    const float* mask = (const float*)d_in[1];
    const float* qkvw = (const float*)d_in[2];
    const float* qkvb = (const float*)d_in[3];
    const float* outw = (const float*)d_in[4];
    const float* outb = (const float*)d_in[5];
    const float* lng  = (const float*)d_in[6];
    const float* lnb  = (const float*)d_in[7];
    float* out = (float*)d_out;
    char* ws = (char*)d_ws;

    _Float16* Xh    = (_Float16*)(ws + XH_OFF);
    _Float16* Wqkvt = (_Float16*)(ws + WQKVT_OFF);
    _Float16* Woutt = (_Float16*)(ws + WOUTT_OFF);
    _Float16* Qh    = (_Float16*)(ws + QH_OFF);
    _Float16* Kh    = (_Float16*)(ws + KH_OFF);
    _Float16* Vt    = (_Float16*)(ws + VT_OFF);
    _Float16* Ctxh  = (_Float16*)(ws + CTX_OFF);
    float*    Resid = (float*)(ws + RESID_OFF);

    const int nX = MROWS * DD;  // 3145728
    cvt_x_kernel<<<nX / 256, 256, 0, stream>>>(x, Xh, nX);

    const float wids[4] = {0.0f, 0.1f, 1.0f, 5.0f};
    for (int br = 0; br < 4; ++br) {
        cvt_w_kernel<<<(4 * DSQ) / 256, 256, 0, stream>>>(
            qkvw + br * 3 * DSQ, outw + br * DSQ, Wqkvt, Woutt);

        dim3 g1(MROWS / 128, 3 * (DD / 64));  // (32, 36)
        qkv_gemm_kernel<<<g1, 256, 0, stream>>>(Xh, Wqkvt, qkvb + br * 3 * DD,
                                                Qh, Kh, Vt);

        float wid = (br == 0) ? 0.01f : wids[br];
        float wdt = ((wid < 0.01f) ? 0.01f : wid) / 9.0f;
        float inv2w2 = 1.0f / (2.0f * wdt * wdt);
        dim3 g2(LSEQ / 16, NB * HH);  // (64, 48)
        attn_kernel<<<g2, 256, 0, stream>>>(Qh, Kh, Vt, mask, Ctxh, inv2w2,
                                            br == 0 ? 0 : 1);

        dim3 g3(MROWS / 128, DD / 64);  // (32, 12)
        out_gemm_kernel<<<g3, 256, 0, stream>>>(Ctxh, Woutt, outb + br * DD, x,
                                                Resid);

        ln_kernel<<<MROWS, 256, 0, stream>>>(Resid, lng + br * DD, lnb + br * DD,
                                             out, br == 0 ? 1 : 0);
    }
}
